// MistralAttention_73993696576088
// MI455X (gfx1250) — compile-verified, matrix operand order checked by reading, not run on hardware
//
#include <hip/hip_runtime.h>
#include <hip/hip_bf16.h>

typedef _Float16 h16;
typedef __attribute__((ext_vector_type(16))) _Float16 v16h;
typedef __attribute__((ext_vector_type(8)))  _Float16 v8h;
typedef __attribute__((ext_vector_type(8)))  float    v8f;

#define S_LEN   2048
#define H_DIM   4096
#define N_HEADS 32
#define N_KV    8
#define HEAD_D  128

// ---------------------------------------------------------------------------
// Fragment loader for 16-bit WMMA operands (A rows / B columns from a
// row-major, K-contiguous matrix).  Per CDNA5 ISA layout:
//   lane l: row/col = l&15 ; halves 0..7  = K (sel*8 .. sel*8+7),
//                            halves 8..15 = K (16+sel*8 ..), sel = l>>4.
// Compiles to two *_load_b128 (global or ds depending on pointer).
// ---------------------------------------------------------------------------
__device__ __forceinline__ v16h load_frag(const h16* __restrict__ base, int stride,
                                          int row0, int k0) {
  const int l   = threadIdx.x & 31;
  const int m   = l & 15;
  const int sel = l >> 4;
  const h16* p  = base + (size_t)(row0 + m) * (size_t)stride + k0 + sel * 8;
  v8h lo = *(const v8h*)(p);
  v8h hi = *(const v8h*)(p + 16);
  v16h r;
#pragma unroll
  for (int i = 0; i < 8; ++i) { r[i] = lo[i]; r[i + 8] = hi[i]; }
  return r;
}

__device__ __forceinline__ v8f wmma16(v16h a, v16h b, v8f c) {
  return __builtin_amdgcn_wmma_f32_16x16x32_f16(false, a, false, b, (short)0, c,
                                                false, false);
}

// ---------------------------------------------------------------------------
// f32 -> f16 conversion (grid-stride)
// ---------------------------------------------------------------------------
__global__ void cvt_f32_f16(const float* __restrict__ src, h16* __restrict__ dst, int n) {
  for (int i = blockIdx.x * blockDim.x + threadIdx.x; i < n; i += gridDim.x * blockDim.x)
    dst[i] = (h16)src[i];
}

// ---------------------------------------------------------------------------
// Block-tiled C = A(MxK) @ B(NxK)^T, both row-major K-contiguous.
// 256-thread block -> 128x128 C tile.  Double-buffered LDS staging via
// gfx1250 async-to-LDS loads: while 8 waves run 8 WMMAs each on buffer b, the
// ASYNCcnt engine fills buffer b^1 (A 128x32 + B 128x32, 8KB each).
// OUT_F16 is a compile-time switch so the store epilogue is branch-free.
// ---------------------------------------------------------------------------
#define BLK_K 32
#define LDS_BUF_BYTES (128 * BLK_K * 2)               // 8KB per matrix per buffer

template <bool OUT_F16>
__global__ void gemm_blk(const h16* __restrict__ A, const h16* __restrict__ B,
                         h16* __restrict__ Cf16, float* __restrict__ Cf32,
                         int M, int N, int K) {
  __shared__ __align__(16) h16 ldsA[2][128 * BLK_K];  // 16KB
  __shared__ __align__(16) h16 ldsB[2][128 * BLK_K];  // 16KB
  const int tid = threadIdx.x;
  const int nbn = N >> 7;
  const int bm  = blockIdx.x / nbn;
  const int bn  = blockIdx.x % nbn;
  const int row0 = bm << 7;
  const int col0 = bn << 7;

  // Staging assignment: chunk c (0..511) = 16B segment; row = c>>2, seg = c&3.
  // Each thread owns chunks 2*tid and 2*tid+1 of both tiles.
  const int c0  = tid * 2;
  const int rA0 = c0 >> 2,       sA0 = c0 & 3;
  const int rA1 = (c0 + 1) >> 2, sA1 = (c0 + 1) & 3;

  // LDS byte offsets (generic LDS pointer low 32 bits = LDS offset)
  const unsigned la0 = (unsigned)(unsigned long long)(void*)&ldsA[0][rA0 * BLK_K + sA0 * 8];
  const unsigned la1 = (unsigned)(unsigned long long)(void*)&ldsA[0][rA1 * BLK_K + sA1 * 8];
  const unsigned lb0 = (unsigned)(unsigned long long)(void*)&ldsB[0][rA0 * BLK_K + sA0 * 8];
  const unsigned lb1 = (unsigned)(unsigned long long)(void*)&ldsB[0][rA1 * BLK_K + sA1 * 8];

  // Per-thread global byte offsets at k0 = 0 (advance 64B per K-chunk)
  unsigned gA0 = ((unsigned)(row0 + rA0) * (unsigned)K + sA0 * 8u) * 2u;
  unsigned gA1 = ((unsigned)(row0 + rA1) * (unsigned)K + sA1 * 8u) * 2u;
  unsigned gB0 = ((unsigned)(col0 + rA0) * (unsigned)K + sA0 * 8u) * 2u;
  unsigned gB1 = ((unsigned)(col0 + rA1) * (unsigned)K + sA1 * 8u) * 2u;

  auto issue_stage = [&](unsigned bufoff) {
    asm volatile("global_load_async_to_lds_b128 %0, %1, %2"
                 :: "v"(la0 + bufoff), "v"(gA0), "s"(A) : "memory");
    asm volatile("global_load_async_to_lds_b128 %0, %1, %2"
                 :: "v"(la1 + bufoff), "v"(gA1), "s"(A) : "memory");
    asm volatile("global_load_async_to_lds_b128 %0, %1, %2"
                 :: "v"(lb0 + bufoff), "v"(gB0), "s"(B) : "memory");
    asm volatile("global_load_async_to_lds_b128 %0, %1, %2"
                 :: "v"(lb1 + bufoff), "v"(gB1), "s"(B) : "memory");
    gA0 += BLK_K * 2; gA1 += BLK_K * 2; gB0 += BLK_K * 2; gB1 += BLK_K * 2;
  };

  // Wave sub-tile: 4 (M) x 2 (N) wave grid; each wave 32 rows x 64 cols.
  const int wv = tid >> 5;
  const int wm = (wv >> 1) << 5;
  const int wn = (wv & 1) << 6;

  v8f zero = {};
  v8f acc[2][4];
#pragma unroll
  for (int i = 0; i < 2; ++i)
#pragma unroll
    for (int j = 0; j < 4; ++j) acc[i][j] = zero;

  unsigned buf = 0;
  issue_stage(0);                                     // prologue: fill buffer 0
  for (int k0 = 0; k0 < K; k0 += BLK_K) {
    asm volatile("s_wait_asynccnt 0" ::: "memory");   // current buffer landed
    __syncthreads();
    if (k0 + BLK_K < K)                               // overlap: fill other buf
      issue_stage((buf ^ 1u) * LDS_BUF_BYTES);

    // ---- compute from LDS fragments (buffer `buf`) ----
    const h16* sA = ldsA[buf];
    const h16* sB = ldsB[buf];
    v16h a0 = load_frag(sA, BLK_K, wm, 0);
    v16h a1 = load_frag(sA, BLK_K, wm + 16, 0);
#pragma unroll
    for (int j = 0; j < 4; ++j) {
      v16h b = load_frag(sB, BLK_K, wn + j * 16, 0);
      acc[0][j] = wmma16(a0, b, acc[0][j]);
      acc[1][j] = wmma16(a1, b, acc[1][j]);
    }

    asm volatile("s_wait_dscnt 0" ::: "memory");      // reads done before reuse
    __syncthreads();
    buf ^= 1u;
  }

  const int l  = threadIdx.x & 31;
  const int nn = l & 15;
  const int hl = l >> 4;
#pragma unroll
  for (int i = 0; i < 2; ++i)
#pragma unroll
    for (int j = 0; j < 4; ++j)
#pragma unroll
      for (int r = 0; r < 8; ++r) {
        const int m = wm + i * 16 + r + hl * 8;       // C layout: VGPR r -> M=r/r+8
        const size_t idx = (size_t)(row0 + m) * N + col0 + wn + j * 16 + nn;
        if constexpr (OUT_F16) Cf16[idx] = (h16)acc[i][j][r];
        else                   Cf32[idx] = acc[i][j][r];
      }
}

// ---------------------------------------------------------------------------
// RoPE + reorder [s][h*128+d] -> [h][s][d] (f16). cos/sin are f32 (1,S,128).
// ---------------------------------------------------------------------------
__global__ void rope_reorder(const h16* __restrict__ Xflat, const float* __restrict__ cosp,
                             const float* __restrict__ sinp, h16* __restrict__ Xh,
                             int nheads) {
  const int total = nheads * S_LEN * HEAD_D;
  const int rowst = nheads * HEAD_D;
  for (int idx = blockIdx.x * blockDim.x + threadIdx.x; idx < total;
       idx += gridDim.x * blockDim.x) {
    const int d = idx & (HEAD_D - 1);
    const int s = (idx >> 7) & (S_LEN - 1);
    const int h = idx >> 18;                          // / (2048*128)
    const float c  = cosp[s * HEAD_D + d];
    const float sn = sinp[s * HEAD_D + d];
    const size_t rb = (size_t)s * rowst + h * HEAD_D;
    const float x = (float)Xflat[rb + d];
    const float o = (d < 64) ? -(float)Xflat[rb + d + 64]
                             :  (float)Xflat[rb + d - 64];
    Xh[idx] = (h16)(x * c + o * sn);
  }
}

// V: [s][kv*128+d] -> Vt [kv][d][s]  (so PV B-fragments are s-contiguous)
__global__ void v_transpose(const h16* __restrict__ Vflat, h16* __restrict__ Vt) {
  const int total = N_KV * S_LEN * HEAD_D;
  for (int idx = blockIdx.x * blockDim.x + threadIdx.x; idx < total;
       idx += gridDim.x * blockDim.x) {
    const int s  = idx & (S_LEN - 1);
    const int d  = (idx >> 11) & (HEAD_D - 1);
    const int kv = idx >> 18;
    Vt[idx] = Vflat[(size_t)s * (N_KV * HEAD_D) + kv * HEAD_D + d];
  }
}

// ---------------------------------------------------------------------------
// Flash attention: one wave per (head, 16-query tile), streaming 32-key blocks.
// Online softmax; causal mask computed analytically (== reference's -1e9 mask).
// __launch_bounds__(256, 1): keep the ~180-VGPR working set (64 acc + 32 Q
// fragments + softmax state) fully resident — no scratch spills.
// ---------------------------------------------------------------------------
__global__ void __launch_bounds__(256, 1)
attn_fwd(const h16* __restrict__ Qh, const h16* __restrict__ Kh,
         const h16* __restrict__ Vt, h16* __restrict__ Oflat) {
  __shared__ __align__(16) h16 plds[8][16 * 32];      // per-wave P staging tile
  const int w   = threadIdx.x >> 5;
  const int wid = blockIdx.x * 8 + w;
  const int h   = wid >> 7;                           // 128 q-tiles per head
  const int qt  = wid & 127;
  const int q0  = qt << 4;
  const int kvh = h >> 2;                             // G = 32/8 = 4
  const int l   = threadIdx.x & 31;
  const int nn  = l & 15;
  const int hl  = l >> 4;
  const float scale = 0.08838834764831845f;           // 1/sqrt(128)

  const h16* qbase = Qh + (size_t)h   * S_LEN * HEAD_D;
  const h16* kbase = Kh + (size_t)kvh * S_LEN * HEAD_D;
  const h16* vbase = Vt + (size_t)kvh * HEAD_D * S_LEN;

  v16h qf[4];
#pragma unroll
  for (int c = 0; c < 4; ++c) qf[c] = load_frag(qbase, HEAD_D, q0, c * 32);

  v8f zero = {};
  v8f acc[8];                                         // 16 x 128 f32 output tile
#pragma unroll
  for (int t = 0; t < 8; ++t) acc[t] = zero;
  float row_max[8], row_sum[8];
#pragma unroll
  for (int r = 0; r < 8; ++r) { row_max[r] = -3.0e38f; row_sum[r] = 0.0f; }

  const int nblk = (q0 + 47) >> 5;                    // keys <= q0+15, blocks of 32
  for (int b = 0; b < nblk; ++b) {
    const int kb = b << 5;

    // S = Q K^T : two 16x16 tiles, 4 WMMAs each over D=128
    v8f st[2];
#pragma unroll
    for (int t = 0; t < 2; ++t) {
      st[t] = zero;
#pragma unroll
      for (int c = 0; c < 4; ++c) {
        v16h kf = load_frag(kbase, HEAD_D, kb + t * 16, c * 32);
        st[t] = wmma16(qf[c], kf, st[t]);
      }
    }

    // online softmax (row stats per C-layout row r + hl*8)
    float scf[8];
#pragma unroll
    for (int r = 0; r < 8; ++r) {
      const int qrow = q0 + r + hl * 8;
      float v0 = st[0][r] * scale;
      float v1 = st[1][r] * scale;
      if (kb + nn > qrow)      v0 = -1.0e30f;
      if (kb + 16 + nn > qrow) v1 = -1.0e30f;
      float bm = fmaxf(v0, v1);
#pragma unroll
      for (int mk = 1; mk < 16; mk <<= 1) bm = fmaxf(bm, __shfl_xor(bm, mk, 32));
      const float nm  = fmaxf(row_max[r], bm);
      const float scv = __expf(row_max[r] - nm);
      row_max[r] = nm;
      const float p0 = __expf(v0 - nm);
      const float p1 = __expf(v1 - nm);
      float bs = p0 + p1;
#pragma unroll
      for (int mk = 1; mk < 16; mk <<= 1) bs += __shfl_xor(bs, mk, 32);
      row_sum[r] = row_sum[r] * scv + bs;
      scf[r] = scv;
      const int m = r + hl * 8;
      plds[w][m * 32 + nn]      = (h16)p0;            // re-layout C->A via LDS
      plds[w][m * 32 + 16 + nn] = (h16)p1;
    }
#pragma unroll
    for (int t = 0; t < 8; ++t)
#pragma unroll
      for (int r = 0; r < 8; ++r) acc[t][r] *= scf[r];

    // same-wave LDS is in-order, but fence the compiler + counters explicitly
    asm volatile("s_wait_dscnt 0" ::: "memory");

    v16h pf;                                          // P as A-fragment (16x32)
    {
      const h16* p = &plds[w][nn * 32 + hl * 8];
      v8h lo = *(const v8h*)(p);
      v8h hi = *(const v8h*)(p + 16);
#pragma unroll
      for (int i = 0; i < 8; ++i) { pf[i] = lo[i]; pf[i + 8] = hi[i]; }
    }

    // O += P V : 8 WMMAs (one per 16-wide d chunk), B from Vt (s-contiguous)
#pragma unroll
    for (int t = 0; t < 8; ++t) {
      v16h vf = load_frag(vbase, S_LEN, t * 16, kb);
      acc[t] = wmma16(pf, vf, acc[t]);
    }
  }

  float inv[8];
#pragma unroll
  for (int r = 0; r < 8; ++r) inv[r] = 1.0f / row_sum[r];
#pragma unroll
  for (int t = 0; t < 8; ++t)
#pragma unroll
    for (int r = 0; r < 8; ++r) {
      const int m = r + hl * 8;
      Oflat[(size_t)(q0 + m) * (N_HEADS * HEAD_D) + h * HEAD_D + t * 16 + nn] =
          (h16)(acc[t][r] * inv[r]);
    }
}

// ---------------------------------------------------------------------------
extern "C" void kernel_launch(void* const* d_in, const int* in_sizes, int n_in,
                              void* d_out, int out_size, void* d_ws, size_t ws_size,
                              hipStream_t stream) {
  const float* hs = (const float*)d_in[0];
  const float* cp = (const float*)d_in[1];
  const float* sp = (const float*)d_in[2];
  // d_in[3] = attention_mask (pure causal; applied analytically in attn_fwd)
  const float* wq = (const float*)d_in[4];
  const float* wk = (const float*)d_in[5];
  const float* wv = (const float*)d_in[6];
  const float* wo = (const float*)d_in[7];
  float* out = (float*)d_out;

  const size_t nX  = (size_t)S_LEN * H_DIM;                 // 8M
  const size_t nWq = (size_t)(N_HEADS * HEAD_D) * H_DIM;    // 16M
  const size_t nWk = (size_t)(N_KV   * HEAD_D) * H_DIM;     // 4M
  const size_t nKV = (size_t)S_LEN * N_KV * HEAD_D;         // 2M

  char* ws = (char*)d_ws;
  h16* XH  = (h16*)ws; ws += nX  * 2;
  h16* WQH = (h16*)ws; ws += nWq * 2;
  h16* WKH = (h16*)ws; ws += nWk * 2;
  h16* WVH = (h16*)ws; ws += nWk * 2;
  h16* WOH = (h16*)ws; ws += nWq * 2;
  h16* QF  = (h16*)ws; ws += nX  * 2;
  h16* KF  = (h16*)ws; ws += nKV * 2;
  h16* VF  = (h16*)ws; ws += nKV * 2;
  h16* QH2 = (h16*)ws; ws += nX  * 2;
  h16* KH2 = (h16*)ws; ws += nKV * 2;
  h16* VT  = (h16*)ws; ws += nKV * 2;
  h16* OF  = (h16*)ws;                                      // total ~160 MB

  cvt_f32_f16<<<2048, 256, 0, stream>>>(hs, XH,  (int)nX);
  cvt_f32_f16<<<2048, 256, 0, stream>>>(wq, WQH, (int)nWq);
  cvt_f32_f16<<<2048, 256, 0, stream>>>(wk, WKH, (int)nWk);
  cvt_f32_f16<<<2048, 256, 0, stream>>>(wv, WVH, (int)nWk);
  cvt_f32_f16<<<2048, 256, 0, stream>>>(wo, WOH, (int)nWq);

  // QKV projections (f16 WMMA, f32 accumulate) — double-buffered async-LDS GEMM
  gemm_blk<true ><<<16 * 32, 256, 0, stream>>>(XH, WQH, QF, nullptr, 2048, 4096, 4096);
  gemm_blk<true ><<<16 *  8, 256, 0, stream>>>(XH, WKH, KF, nullptr, 2048, 1024, 4096);
  gemm_blk<true ><<<16 *  8, 256, 0, stream>>>(XH, WVH, VF, nullptr, 2048, 1024, 4096);

  rope_reorder<<<4096, 256, 0, stream>>>(QF, cp, sp, QH2, N_HEADS);
  rope_reorder<<<4096, 256, 0, stream>>>(KF, cp, sp, KH2, N_KV);
  v_transpose<<<2048, 256, 0, stream>>>(VF, VT);

  attn_fwd<<<(N_HEADS * 128) / 8, 256, 0, stream>>>(QH2, KH2, VT, OF);

  // output projection -> f32 d_out
  gemm_blk<false><<<16 * 32, 256, 0, stream>>>(OF, WOH, nullptr, out, 2048, 4096, 4096);
}
